// Lilly_6734508720583
// MI455X (gfx1250) — matrix operand order repaired and checked by reading, not compiled, not run on hardware
//
#include <hip/hip_runtime.h>
#include <hip/hip_bf16.h>
#include <math.h>

typedef __bf16 v16bf __attribute__((ext_vector_type(16)));
typedef float  v8f   __attribute__((ext_vector_type(8)));

#define TOKENS 2048   // B*S = 4*512
#define EDIM   512
#define FDIM   2048
#define VOCAB  50257
#define NHEAD  8
#define HDIM   64
#define LB     4      // attention "sequence" length (batch axis)
#define NN     512    // attention "batch" (seq axis)

// ---------------------------------------------------------------------------
// f32 -> bf16 elementwise convert (weight staging / activation copies)
// ---------------------------------------------------------------------------
__global__ __launch_bounds__(256) void cvt_bf16_kernel(
    const float* __restrict__ s, __bf16* __restrict__ d, int n) {
  int i = (blockIdx.x * 256 + threadIdx.x) * 4;
  if (i + 3 < n) {
    float4 v = *(const float4*)(s + i);
    d[i]     = (__bf16)v.x;
    d[i + 1] = (__bf16)v.y;
    d[i + 2] = (__bf16)v.z;
    d[i + 3] = (__bf16)v.w;
  } else {
    for (; i < n; ++i) d[i] = (__bf16)s[i];
  }
}

// ---------------------------------------------------------------------------
// Embedding: sin table for num tokens, learned gather otherwise, * sqrt(E),
// plus sinusoidal positional encoding.  Writes f32 (residual) + bf16 (GEMM A).
// ---------------------------------------------------------------------------
__global__ __launch_bounds__(256) void embed_kernel(
    const int* __restrict__ x, const float* __restrict__ emb_w,
    float* __restrict__ h, __bf16* __restrict__ hbf) {
  int idx = blockIdx.x * blockDim.x + threadIdx.x;       // [0, TOKENS*EDIM)
  if (idx >= TOKENS * EDIM) return;
  int row = idx >> 9;            // token row (b*S + s)
  int e   = idx & 511;
  int s   = row & 511;           // S index (PE position)
  int t   = x[row];
  float v;
  if (t < 1000) {
    v = sinf(((float)t * 0.001f) * (float)(e + 1));
  } else {
    v = emb_w[(size_t)t * EDIM + e];
  }
  v *= 22.62741699796952f;       // sqrt(512)
  float ang = (float)s * expf(-(float)(e & ~1) * (9.210340371976184f / 512.0f));
  v += (e & 1) ? cosf(ang) : sinf(ang);
  h[idx]   = v;
  hbf[idx] = (__bf16)v;
}

// ---------------------------------------------------------------------------
// Generic GEMM: C[M,N] = act(A[M,K] * W[N,K]^T + bias[N])
// Pure-bf16 inputs: fragments are contiguous bf16 runs that load directly as
// b128 into the WMMA source registers (no cvt, no shuffle in the K-loop).
// Block: 256 threads = 8 waves; block tile 128(M) x 128(N);
// wave tile 64(M) x 32(N) = 4x2 16x16 f32 accumulators.  M % 128 == 0.
// N tail: weight row index clamped to N-1; garbage columns never stored.
// ---------------------------------------------------------------------------
template <int RELU, typename TC>
__global__ __launch_bounds__(256) void gemm_wmma_bf16(
    const __bf16* __restrict__ A, const __bf16* __restrict__ W,
    const float* __restrict__ bias, TC* __restrict__ C,
    int M, int N, int K) {
  const int lane = threadIdx.x & 31;
  const int wave = threadIdx.x >> 5;
  const int l15  = lane & 15;
  const int lh   = lane >> 4;                 // lane half: 0 or 1

  const int m0 = blockIdx.y * 128 + (wave >> 2) * 64;  // wave M origin
  const int n0 = blockIdx.x * 128 + (wave & 3) * 32;   // wave N origin

  const __bf16* wrow[2];
#pragma unroll
  for (int ni = 0; ni < 2; ++ni) {
    int n = n0 + ni * 16 + l15;
    int nc = (n < N) ? n : (N - 1);
    wrow[ni] = W + (size_t)nc * K;
  }
  const __bf16* arow[4];
#pragma unroll
  for (int mi = 0; mi < 4; ++mi)
    arow[mi] = A + (size_t)(m0 + mi * 16 + l15) * K;

  const v8f vzero = {0.f, 0.f, 0.f, 0.f, 0.f, 0.f, 0.f, 0.f};
  v8f acc[4][2];
#pragma unroll
  for (int mi = 0; mi < 4; ++mi)
#pragma unroll
    for (int ni = 0; ni < 2; ++ni) acc[mi][ni] = vzero;

  for (int k0 = 0; k0 < K; k0 += 32) {
    // --- A fragments: 16-bit A 16x32 layout.
    // Elements 0..7  = K = lh*8      .. lh*8+7   (contiguous bf16, 16 B)
    // Elements 8..15 = K = 16 + lh*8 .. +7       (contiguous bf16, 16 B)
    v16bf afr[4];
#pragma unroll
    for (int mi = 0; mi < 4; ++mi) {
      const __bf16* ar = arow[mi] + k0;
#pragma unroll
      for (int j = 0; j < 8; ++j) {
        int k = ((j < 4) ? 0 : 16) + lh * 8 + 2 * (j & 3);
        afr[mi][2 * j]     = ar[k];
        afr[mi][2 * j + 1] = ar[k + 1];
      }
    }
    // --- B fragments: 16-bit B 32x16 layout (B = W^T, column n = W row n).
    // Elements 0..7 = K = lh*8 .. lh*8+7; elements 8..15 = K = 16 + lh*8 .. +7
    v16bf bfr[2];
#pragma unroll
    for (int ni = 0; ni < 2; ++ni) {
      const __bf16* wr = wrow[ni] + k0;
#pragma unroll
      for (int j = 0; j < 8; ++j) {
        int k = ((j < 4) ? 0 : 16) + lh * 8 + 2 * (j & 3);
        bfr[ni][2 * j]     = wr[k];
        bfr[ni][2 * j + 1] = wr[k + 1];
      }
    }
#pragma unroll
    for (int mi = 0; mi < 4; ++mi)
#pragma unroll
      for (int ni = 0; ni < 2; ++ni)
        acc[mi][ni] = __builtin_amdgcn_wmma_f32_16x16x32_bf16(
            false, afr[mi], false, bfr[ni], (short)0, acc[mi][ni], false, false);
  }

  // --- store: C/D layout: VGPR r -> M = r + lh*8, N = l15
#pragma unroll
  for (int ni = 0; ni < 2; ++ni) {
    int n = n0 + ni * 16 + l15;
    if (n < N) {
      float bv = bias[n];
#pragma unroll
      for (int mi = 0; mi < 4; ++mi) {
#pragma unroll
        for (int r = 0; r < 8; ++r) {
          int m = m0 + mi * 16 + lh * 8 + r;
          float v = acc[mi][ni][r] + bv;
          if (RELU) v = fmaxf(v, 0.f);
          C[(size_t)m * N + n] = (TC)v;
        }
      }
    }
  }
}

// ---------------------------------------------------------------------------
// Attention over axis 0 (length 4): one thread per (n, head).
// qkv layout [l, n, 3E]: q at f, k at 512+f, v at 1024+f  (f = h*64 + d).
// Output written bf16 (it is only consumed as the A matrix of out-proj).
// ---------------------------------------------------------------------------
__global__ __launch_bounds__(128) void attn_kernel(
    const float* __restrict__ qkv, __bf16* __restrict__ out) {
  int idx = blockIdx.x * blockDim.x + threadIdx.x;
  if (idx >= NN * NHEAD) return;
  int n = idx >> 3;
  int h = idx & 7;
  size_t base[LB];
#pragma unroll
  for (int l = 0; l < LB; ++l)
    base[l] = (size_t)(l * NN + n) * (3 * EDIM) + h * HDIM;

  float s[LB][LB];
#pragma unroll
  for (int l = 0; l < LB; ++l) {
#pragma unroll
    for (int m = 0; m < LB; ++m) {
      float a = 0.f;
      for (int d = 0; d < HDIM; ++d)
        a += qkv[base[l] + d] * qkv[base[m] + EDIM + d];
      s[l][m] = a * 0.125f;  // 1/sqrt(64)
    }
  }
#pragma unroll
  for (int l = 0; l < LB; ++l) {
    float mx = s[l][0];
#pragma unroll
    for (int m = 1; m < LB; ++m) mx = fmaxf(mx, s[l][m]);
    float sum = 0.f;
#pragma unroll
    for (int m = 0; m < LB; ++m) { s[l][m] = expf(s[l][m] - mx); sum += s[l][m]; }
    float inv = 1.f / sum;
#pragma unroll
    for (int m = 0; m < LB; ++m) s[l][m] *= inv;
  }
  for (int d = 0; d < HDIM; ++d) {
    float vm[LB];
#pragma unroll
    for (int m = 0; m < LB; ++m) vm[m] = qkv[base[m] + 2 * EDIM + d];
#pragma unroll
    for (int l = 0; l < LB; ++l) {
      float o = s[l][0] * vm[0] + s[l][1] * vm[1] + s[l][2] * vm[2] + s[l][3] * vm[3];
      out[(size_t)(l * NN + n) * EDIM + h * HDIM + d] = (__bf16)o;
    }
  }
}

// ---------------------------------------------------------------------------
// Fused residual + layernorm: y = LN(x + r) * w + b.
// Writes f32 (next residual) and bf16 (next GEMM A).
// ---------------------------------------------------------------------------
__global__ __launch_bounds__(128) void add_ln_kernel(
    const float* __restrict__ x, const float* __restrict__ r,
    const float* __restrict__ w, const float* __restrict__ b,
    float* __restrict__ y, __bf16* __restrict__ ybf) {
  __shared__ float s1[128], s2[128];
  int row = blockIdx.x;
  int tid = threadIdx.x;
  const float* xr = x + (size_t)row * EDIM;
  const float* rr = r + (size_t)row * EDIM;
  float t[4], sum = 0.f, sq = 0.f;
#pragma unroll
  for (int i = 0; i < 4; ++i) {
    int e = tid + i * 128;
    t[i] = xr[e] + rr[e];
    sum += t[i];
    sq  += t[i] * t[i];
  }
  s1[tid] = sum; s2[tid] = sq;
  __syncthreads();
  for (int stride = 64; stride > 0; stride >>= 1) {
    if (tid < stride) { s1[tid] += s1[tid + stride]; s2[tid] += s2[tid + stride]; }
    __syncthreads();
  }
  float mu   = s1[0] * (1.0f / EDIM);
  float var  = s2[0] * (1.0f / EDIM) - mu * mu;
  float rstd = rsqrtf(var + 1e-5f);
#pragma unroll
  for (int i = 0; i < 4; ++i) {
    int e = tid + i * 128;
    float v = (t[i] - mu) * rstd * w[e] + b[e];
    y[(size_t)row * EDIM + e]   = v;
    ybf[(size_t)row * EDIM + e] = (__bf16)v;
  }
}

// ---------------------------------------------------------------------------
// Orchestration
// ---------------------------------------------------------------------------
static inline void cvt_w(const float* src, __bf16* dst, int n, hipStream_t stream) {
  cvt_bf16_kernel<<<(n / 4 + 255) / 256, 256, 0, stream>>>(src, dst, n);
}

extern "C" void kernel_launch(void* const* d_in, const int* in_sizes, int n_in,
                              void* d_out, int out_size, void* d_ws, size_t ws_size,
                              hipStream_t stream) {
  const int*   x          = (const int*)d_in[0];
  const float* emb_w      = (const float*)d_in[1];
  const float* in_proj_w  = (const float*)d_in[2];
  const float* in_proj_b  = (const float*)d_in[3];
  const float* out_proj_w = (const float*)d_in[4];
  const float* out_proj_b = (const float*)d_in[5];
  const float* ln1_w      = (const float*)d_in[6];
  const float* ln1_b      = (const float*)d_in[7];
  const float* ln2_w      = (const float*)d_in[8];
  const float* ln2_b      = (const float*)d_in[9];
  const float* ffn_w1     = (const float*)d_in[10];
  const float* ffn_b1     = (const float*)d_in[11];
  const float* ffn_w2     = (const float*)d_in[12];
  const float* ffn_b2     = (const float*)d_in[13];
  const float* dec_w      = (const float*)d_in[14];
  const float* dec_b      = (const float*)d_in[15];
  float* out = (float*)d_out;

  // ---- scratch layout ----
  // f32 region: h0, h2, qkv, proj, ffo
  float* ws   = (float*)d_ws;
  float* h0   = ws;                          // 2048*512
  float* h2   = h0 + TOKENS * EDIM;          // 2048*512
  float* qkv  = h2 + TOKENS * EDIM;          // 2048*1536
  float* proj = qkv + TOKENS * 3 * EDIM;     // 2048*512
  float* ffo  = proj + TOKENS * EDIM;        // 2048*512
  // bf16 region
  __bf16* h0bf   = (__bf16*)(ffo + TOKENS * EDIM);
  __bf16* h2bf   = h0bf + TOKENS * EDIM;     // 2048*512
  __bf16* attnbf = h2bf + TOKENS * EDIM;     // 2048*512
  __bf16* midbf  = attnbf + TOKENS * EDIM;   // 2048*2048
  __bf16* wbf    = midbf + TOKENS * FDIM;    // up to VOCAB*EDIM (decoder weight)

  // 1) embedding + positional encoding
  embed_kernel<<<(TOKENS * EDIM + 255) / 256, 256, 0, stream>>>(x, emb_w, h0, h0bf);

  const dim3 blk(256);
  for (int l = 0; l < 2; ++l) {
    // qkv = h0 @ in_proj_w[l]^T + b   [2048 x 1536]
    cvt_w(in_proj_w + (size_t)l * 3 * EDIM * EDIM, wbf, 3 * EDIM * EDIM, stream);
    gemm_wmma_bf16<0, float><<<dim3((3 * EDIM) / 128, TOKENS / 128), blk, 0, stream>>>(
        h0bf, wbf, in_proj_b + (size_t)l * 3 * EDIM, qkv, TOKENS, 3 * EDIM, EDIM);
    // attention (length-4 axis) -> bf16
    attn_kernel<<<(NN * NHEAD + 127) / 128, 128, 0, stream>>>(qkv, attnbf);
    // out projection
    cvt_w(out_proj_w + (size_t)l * EDIM * EDIM, wbf, EDIM * EDIM, stream);
    gemm_wmma_bf16<0, float><<<dim3(EDIM / 128, TOKENS / 128), blk, 0, stream>>>(
        attnbf, wbf, out_proj_b + (size_t)l * EDIM, proj, TOKENS, EDIM, EDIM);
    // h2 = LN(h0 + proj)
    add_ln_kernel<<<TOKENS, 128, 0, stream>>>(h0, proj,
        ln1_w + (size_t)l * EDIM, ln1_b + (size_t)l * EDIM, h2, h2bf);
    // mid = relu(h2 @ ffn_w1^T + b1)   [2048 x 2048], bf16-only output
    cvt_w(ffn_w1 + (size_t)l * FDIM * EDIM, wbf, FDIM * EDIM, stream);
    gemm_wmma_bf16<1, __bf16><<<dim3(FDIM / 128, TOKENS / 128), blk, 0, stream>>>(
        h2bf, wbf, ffn_b1 + (size_t)l * FDIM, midbf, TOKENS, FDIM, EDIM);
    // ffo = mid @ ffn_w2^T + b2        [2048 x 512], K = 2048
    cvt_w(ffn_w2 + (size_t)l * EDIM * FDIM, wbf, EDIM * FDIM, stream);
    gemm_wmma_bf16<0, float><<<dim3(EDIM / 128, TOKENS / 128), blk, 0, stream>>>(
        midbf, wbf, ffn_b2 + (size_t)l * EDIM, ffo, TOKENS, EDIM, FDIM);
    // h0 = LN(h2 + ffo)
    add_ln_kernel<<<TOKENS, 128, 0, stream>>>(h2, ffo,
        ln2_w + (size_t)l * EDIM, ln2_b + (size_t)l * EDIM, h0, h0bf);
  }

  // 3) decoder: out = h0 @ dec_w^T + dec_b  [2048 x 50257]
  cvt_w(dec_w, wbf, VOCAB * EDIM, stream);
  gemm_wmma_bf16<0, float><<<dim3((VOCAB + 127) / 128, TOKENS / 128), blk, 0, stream>>>(
      h0bf, wbf, dec_b, out, TOKENS, VOCAB, EDIM);
}
